// SelfAttention_37211596653229
// MI455X (gfx1250) — matrix operand order repaired and checked by reading, not compiled, not run on hardware
//
#include <hip/hip_runtime.h>

// ---------------------------------------------------------------------------
// Self-attention for MI455X (gfx1250), bf16 WMMA pipeline, 64x64 register
// blocking per wave (4x4 grid of v_wmma_f32_16x16x32_bf16 tiles).
//   B=4, S=2048, HIDDEN=1024, INNER=1024, scale = INNER^-0.5 = 1/32
// ---------------------------------------------------------------------------

#define BATCH 4
#define SEQ   2048
#define HID   1024          // K-dim of projections
#define INR   1024          // N-dim of projections / head dim
#define ROWS  (BATCH * SEQ) // 8192 flattened rows

typedef unsigned short ushort_t;                       // bf16 bit storage
typedef __bf16 bf16e;
typedef bf16e v16bf __attribute__((ext_vector_type(16)));
typedef float v8f   __attribute__((ext_vector_type(8)));

struct alignas(16) U128 { unsigned int w[4]; };
union FragU { v16bf v; U128 h[2]; };

// fp32 -> bf16 round-to-nearest-even (bit pattern)
__device__ __forceinline__ ushort_t f2bf(float f) {
    unsigned int u = __float_as_uint(f);
    u += 0x7FFFu + ((u >> 16) & 1u);
    return (ushort_t)(u >> 16);
}

// A-fragment: 16x32 bf16 tile, source row-major [16 rows][K contiguous], ld elems/row.
// ISA layout: lanes 0-15 row M=lane, K = {0..7, 16..23}; lanes 16-31 row M=lane-16,
// K = {8..15, 24..31}.
__device__ __forceinline__ v16bf load_frag_a(const ushort_t* p, int ld) {
    const int lane = threadIdx.x & 31;
    const int r    = lane & 15;
    const int kh   = (lane >> 4) << 3;     // 0 or 8
    const ushort_t* base = p + (size_t)r * ld + kh;
    FragU f;
    f.h[0] = *(const U128*)(base);         // K = kh .. kh+7
    f.h[1] = *(const U128*)(base + 16);    // K = 16+kh .. 16+kh+7
    return f.v;
}

// B-fragment: 32x16 bf16 tile, source stored "column-major as rows": [16 cols][K contiguous].
// ISA layout: lanes 0-15 col N=lane, K = {0..7, 16..23}; lanes 16-31 K = {8..15, 24..31}.
__device__ __forceinline__ v16bf load_frag_b(const ushort_t* p, int ld) {
    const int lane = threadIdx.x & 31;
    const int n    = lane & 15;
    const int kh   = (lane >> 4) << 3;     // 0 or 8
    const ushort_t* base = p + (size_t)n * ld + kh;
    FragU f;
    f.h[0] = *(const U128*)(base);         // K = kh .. kh+7
    f.h[1] = *(const U128*)(base + 16);    // K = 16+kh .. 16+kh+7
    return f.v;
}

__device__ __forceinline__ v8f wmma_bf16(v16bf a, v16bf b, v8f c) {
    return __builtin_amdgcn_wmma_f32_16x16x32_bf16(
        /*neg_a=*/false, a, /*neg_b=*/false, b,
        /*c_mod=*/(short)0, c, /*reuse_a=*/false, /*reuse_b=*/false);
}

// 64x64 register-blocked wave GEMM: acc[i][j] += A(64xK) * B(Kx64) subtiles.
// arow: row-major [64][K], lda; brow: N-major [64][K], ldb.
template <int KDIM>
__device__ __forceinline__ void wave_gemm_64x64(const ushort_t* __restrict__ arow, int lda,
                                                const ushort_t* __restrict__ brow, int ldb,
                                                v8f acc[4][4]) {
    #pragma unroll 2
    for (int kk = 0; kk < KDIM; kk += 32) {
        __builtin_prefetch(arow + kk + 256, 0, 1);
        __builtin_prefetch(brow + kk + 256, 0, 1);
        v16bf aF[4];
        #pragma unroll
        for (int i = 0; i < 4; ++i)
            aF[i] = load_frag_a(arow + (size_t)(i * 16) * lda + kk, lda);
        #pragma unroll
        for (int j = 0; j < 4; ++j) {
            v16bf bF = load_frag_b(brow + (size_t)(j * 16) * ldb + kk, ldb);
            #pragma unroll
            for (int i = 0; i < 4; ++i)
                acc[i][j] = wmma_bf16(aF[i], bF, acc[i][j]);
        }
    }
}

__device__ __forceinline__ void zero_acc(v8f acc[4][4]) {
    #pragma unroll
    for (int i = 0; i < 4; ++i)
        #pragma unroll
        for (int j = 0; j < 4; ++j) {
            v8f z = {0.f, 0.f, 0.f, 0.f, 0.f, 0.f, 0.f, 0.f};
            acc[i][j] = z;
        }
}

// ---------------------------------------------------------------------------
// Kernel 1: x fp32 -> bf16 (row-major [ROWS][HID])
// ---------------------------------------------------------------------------
__global__ void cvt_x_kernel(const float* __restrict__ x, ushort_t* __restrict__ xb, int n) {
    int i = blockIdx.x * blockDim.x + threadIdx.x;
    if (i < n) xb[i] = f2bf(x[i]);
}

// ---------------------------------------------------------------------------
// Kernel 2: W[K][N] fp32 -> WT[N][K] bf16, for the 3 weight matrices
// ---------------------------------------------------------------------------
__global__ void cvt_w_kernel(const float* __restrict__ Wq, const float* __restrict__ Wk,
                             const float* __restrict__ Wv,
                             ushort_t* __restrict__ WqT, ushort_t* __restrict__ WkT,
                             ushort_t* __restrict__ WvT) {
    int idx = blockIdx.x * blockDim.x + threadIdx.x;
    const int per = HID * INR;
    if (idx >= 3 * per) return;
    int which = idx / per;
    int rem   = idx - which * per;
    int k = rem / INR;          // row of W
    int n = rem - k * INR;      // col of W
    const float* W = (which == 0) ? Wq : (which == 1) ? Wk : Wv;
    ushort_t*   WT = (which == 0) ? WqT : (which == 1) ? WkT : WvT;
    WT[(size_t)n * HID + k] = f2bf(W[rem]);
}

// ---------------------------------------------------------------------------
// Kernel 3: QKV projection GEMM.  One wave per 64x64 output tile.
//   grid = (ROWS/64, INR/256, 3), block = 128 (4 waves, distinct N tiles)
//   z=0 -> q (row-major bf16), z=1 -> k (row-major bf16), z=2 -> vT[b][n][s] bf16
// ---------------------------------------------------------------------------
__global__ __launch_bounds__(128) void qkv_gemm_kernel(
    const ushort_t* __restrict__ xb,
    const ushort_t* __restrict__ WqT, const ushort_t* __restrict__ WkT,
    const ushort_t* __restrict__ WvT,
    const float* __restrict__ bq, const float* __restrict__ bk,
    const float* __restrict__ bv,
    ushort_t* __restrict__ q, ushort_t* __restrict__ k, ushort_t* __restrict__ vT) {

    const int wave = threadIdx.x >> 5;
    const int lane = threadIdx.x & 31;
    const int z  = blockIdx.z;
    const int m0 = blockIdx.x * 64;                       // row tile base in [0, ROWS)
    const int n0 = (blockIdx.y * 4 + wave) * 64;          // output column tile base

    const ushort_t* WT  = (z == 0) ? WqT : (z == 1) ? WkT : WvT;
    const float*  bias  = (z == 0) ? bq  : (z == 1) ? bk  : bv;

    v8f acc[4][4];
    zero_acc(acc);
    wave_gemm_64x64<HID>(xb + (size_t)m0 * HID, HID, WT + (size_t)n0 * HID, HID, acc);

    // C layout: lane<16 -> (M = e, N = lane); lane>=16 -> (M = 8+e, N = lane-16)
    const int half = (lane >> 4) << 3;                    // 0 or 8
    if (z < 2) {
        ushort_t* out = (z == 0) ? q : k;                 // row-major [ROWS][INR]
        #pragma unroll
        for (int j = 0; j < 4; ++j) {
            const int n = n0 + j * 16 + (lane & 15);
            const float bn = bias[n];
            #pragma unroll
            for (int i = 0; i < 4; ++i) {
                #pragma unroll
                for (int e = 0; e < 8; ++e)
                    out[(size_t)(m0 + i * 16 + half + e) * INR + n] = f2bf(acc[i][j][e] + bn);
            }
        }
    } else {
        // vT[b][n][s]: per-lane 8 contiguous s-values -> one 16B store per subtile
        const int b   = m0 / SEQ;
        const int sb  = m0 % SEQ;
        #pragma unroll
        for (int j = 0; j < 4; ++j) {
            const int n = n0 + j * 16 + (lane & 15);
            const float bn = bias[n];
            #pragma unroll
            for (int i = 0; i < 4; ++i) {
                alignas(16) ushort_t tmp[8];
                #pragma unroll
                for (int e = 0; e < 8; ++e) tmp[e] = f2bf(acc[i][j][e] + bn);
                const int s0 = sb + i * 16 + half;
                *(U128*)(vT + ((size_t)b * INR + n) * SEQ + s0) = *(const U128*)tmp;
            }
        }
    }
}

// ---------------------------------------------------------------------------
// Kernel 4: scores = (q @ k^T) * scale, fp32.  One wave per 64x64 score tile.
//   grid = (ROWS/64, SEQ/256), block = 128 (4 waves)
// ---------------------------------------------------------------------------
__global__ __launch_bounds__(128) void scores_kernel(
    const ushort_t* __restrict__ q, const ushort_t* __restrict__ k,
    float* __restrict__ sc) {

    const int wave = threadIdx.x >> 5;
    const int lane = threadIdx.x & 31;
    const int m0 = blockIdx.x * 64;                       // global row base in [0, ROWS)
    const int n0 = (blockIdx.y * 4 + wave) * 64;          // key column tile base in [0, SEQ)
    const int b  = m0 / SEQ;

    v8f acc[4][4];
    zero_acc(acc);
    wave_gemm_64x64<INR>(q + (size_t)m0 * INR, INR,
                         k + ((size_t)b * SEQ + n0) * INR, INR, acc);  // k rows == k^T cols

    const float scale = 0.03125f;                          // 1024^-0.5
    const int half = (lane >> 4) << 3;
    #pragma unroll
    for (int j = 0; j < 4; ++j) {
        const int n = n0 + j * 16 + (lane & 15);
        #pragma unroll
        for (int i = 0; i < 4; ++i) {
            #pragma unroll
            for (int e = 0; e < 8; ++e)
                sc[(size_t)(m0 + i * 16 + half + e) * SEQ + n] = acc[i][j][e] * scale;
        }
    }
}

// ---------------------------------------------------------------------------
// Kernel 5: row-wise softmax over sc[ROWS][SEQ] (in-place exp), probs -> bf16.
//   grid = ROWS, block = 256
// ---------------------------------------------------------------------------
__global__ __launch_bounds__(256) void softmax_kernel(
    float* __restrict__ sc, ushort_t* __restrict__ pr) {

    const int row = blockIdx.x;
    float* s = sc + (size_t)row * SEQ;
    ushort_t* p = pr + (size_t)row * SEQ;
    __shared__ float red[256];

    float m = -3.0e38f;
    for (int i = threadIdx.x; i < SEQ; i += 256) m = fmaxf(m, s[i]);
    red[threadIdx.x] = m;
    __syncthreads();
    for (int off = 128; off > 0; off >>= 1) {
        if (threadIdx.x < off) red[threadIdx.x] = fmaxf(red[threadIdx.x], red[threadIdx.x + off]);
        __syncthreads();
    }
    m = red[0];
    __syncthreads();

    float sum = 0.f;
    for (int i = threadIdx.x; i < SEQ; i += 256) {
        float e = __expf(s[i] - m);
        s[i] = e;                                          // in-place (regenerated every launch)
        sum += e;
    }
    red[threadIdx.x] = sum;
    __syncthreads();
    for (int off = 128; off > 0; off >>= 1) {
        if (threadIdx.x < off) red[threadIdx.x] += red[threadIdx.x + off];
        __syncthreads();
    }
    const float inv = 1.f / red[0];

    for (int i = threadIdx.x; i < SEQ; i += 256) p[i] = f2bf(s[i] * inv);
}

// ---------------------------------------------------------------------------
// Kernel 6: out = probs @ v   (K = SEQ).  One wave per 64x64 output tile.
//   grid = (ROWS/64, INR/256), block = 128 (4 waves)
// ---------------------------------------------------------------------------
__global__ __launch_bounds__(128) void context_gemm_kernel(
    const ushort_t* __restrict__ pr, const ushort_t* __restrict__ vT,
    float* __restrict__ out) {

    const int wave = threadIdx.x >> 5;
    const int lane = threadIdx.x & 31;
    const int m0 = blockIdx.x * 64;                       // global row base
    const int n0 = (blockIdx.y * 4 + wave) * 64;          // head-dim column tile base
    const int b  = m0 / SEQ;

    v8f acc[4][4];
    zero_acc(acc);
    wave_gemm_64x64<SEQ>(pr + (size_t)m0 * SEQ, SEQ,
                         vT + ((size_t)b * INR + n0) * SEQ, SEQ, acc);

    const int half = (lane >> 4) << 3;
    #pragma unroll
    for (int j = 0; j < 4; ++j) {
        const int n = n0 + j * 16 + (lane & 15);
        #pragma unroll
        for (int i = 0; i < 4; ++i) {
            #pragma unroll
            for (int e = 0; e < 8; ++e)
                out[(size_t)(m0 + i * 16 + half + e) * INR + n] = acc[i][j][e];
        }
    }
}

// ---------------------------------------------------------------------------
// Host-side launch
// ---------------------------------------------------------------------------
extern "C" void kernel_launch(void* const* d_in, const int* in_sizes, int n_in,
                              void* d_out, int out_size, void* d_ws, size_t ws_size,
                              hipStream_t stream) {
    (void)in_sizes; (void)n_in; (void)out_size; (void)ws_size;

    const float* x  = (const float*)d_in[0];
    const float* Wq = (const float*)d_in[1];
    const float* bq = (const float*)d_in[2];
    const float* Wk = (const float*)d_in[3];
    const float* bk = (const float*)d_in[4];
    const float* Wv = (const float*)d_in[5];
    const float* bv = (const float*)d_in[6];
    float* out = (float*)d_out;

    // Workspace layout (bytes)
    char* w = (char*)d_ws;
    ushort_t* xb  = (ushort_t*)(w + 0);                       // 16 MB
    ushort_t* WqT = (ushort_t*)(w + 16777216);                //  2 MB
    ushort_t* WkT = (ushort_t*)(w + 18874368);                //  2 MB
    ushort_t* WvT = (ushort_t*)(w + 20971520);                //  2 MB
    ushort_t* q   = (ushort_t*)(w + 23068672);                // 16 MB
    ushort_t* k   = (ushort_t*)(w + 39845888);                // 16 MB
    ushort_t* vT  = (ushort_t*)(w + 56623104);                // 16 MB
    float*    sc  = (float*)   (w + 73400320);                // 64 MB
    ushort_t* pr  = (ushort_t*)(w + 140509184);               // 32 MB  (total ~166 MB)

    // 1) x -> bf16
    {
        int n = ROWS * HID;
        cvt_x_kernel<<<(n + 255) / 256, 256, 0, stream>>>(x, xb, n);
    }
    // 2) W -> WT bf16
    {
        int n = 3 * HID * INR;
        cvt_w_kernel<<<(n + 255) / 256, 256, 0, stream>>>(Wq, Wk, Wv, WqT, WkT, WvT);
    }
    // 3) QKV projections (WMMA, 64x64 per wave)
    qkv_gemm_kernel<<<dim3(ROWS / 64, INR / 256, 3), 128, 0, stream>>>(
        xb, WqT, WkT, WvT, bq, bk, bv, q, k, vT);
    // 4) scores (WMMA, 64x64 per wave)
    scores_kernel<<<dim3(ROWS / 64, SEQ / 256), 128, 0, stream>>>(q, k, sc);
    // 5) softmax
    softmax_kernel<<<ROWS, 256, 0, stream>>>(sc, pr);
    // 6) context GEMM (WMMA, 64x64 per wave)
    context_gemm_kernel<<<dim3(ROWS / 64, INR / 256), 128, 0, stream>>>(pr, vT, out);
}
